// MessagePassingMapper_38817914421563
// MI455X (gfx1250) — matrix operand order repaired and checked by reading, not compiled, not run on hardware
//
#include <hip/hip_runtime.h>
#include <hip/hip_bf16.h>

typedef __attribute__((ext_vector_type(16))) __bf16 v16bf;
typedef __attribute__((ext_vector_type(8)))  float  v8f;
typedef unsigned short u16;
typedef unsigned int   u32;

#define HID     128
#define EDIM    8
#define NEDGE   640000
#define NSRC    40000
#define NDST    10000
#define NLAYERS 4

// ---------- bf16 <-> f32 (RNE, bit-level) ----------
__device__ __forceinline__ u16 f2bf(float x) {
  u32 u = __float_as_uint(x);
  u32 r = u + 0x7FFFu + ((u >> 16) & 1u);
  return (u16)(r >> 16);
}
__device__ __forceinline__ float bf2f(u16 h) {
  return __uint_as_float(((u32)h) << 16);
}

// fast SiLU: x * rcp(1 + exp(-x))
__device__ __forceinline__ float silu(float x) {
  return x * __builtin_amdgcn_rcpf(1.f + __expf(-x));
}

// hardware f32 atomic add at device scope (L2 RMW), no return (STOREcnt)
__device__ __forceinline__ void atomic_add_f32(float* p, float v) {
  asm volatile("global_atomic_add_f32 %0, %1, off scope:SCOPE_DEV" : : "v"(p), "v"(v) : "memory");
}

// async copy 16B: global -> LDS (ASYNCcnt)
__device__ __forceinline__ void async_cp16(const void* gp, void* lds) {
  u32 la = (u32)(uintptr_t)lds;
  asm volatile("global_load_async_to_lds_b128 %0, %1, off" : : "v"(la), "v"(gp) : "memory");
}
__device__ __forceinline__ void async_wait() {
  asm volatile("s_wait_asynccnt 0x0" : : : "memory");
}

// ---------- WMMA fragment helpers ----------
// blockDim.x == 256 -> 8 waves; wave w owns output columns [16w, 16w+16).
// A (16x32 bf16): per lane two contiguous 16B runs: K[h*8..h*8+7], K[16+h*8..16+h*8+7]
__device__ __forceinline__ v16bf a_frag(const u16* tile, int W, int kc) {
  int lane = threadIdx.x & 31;
  const u16* r = tile + (lane & 15) * W + kc * 32 + ((lane >> 4) << 3);
  union { v16bf v; uint4 u4[2]; } f;
  f.u4[0] = *(const uint4*)(r);
  f.u4[1] = *(const uint4*)(r + 16);
  return f.v;
}

// B fragments pre-packed lane-contiguously by pack_w_kernel: one 32-byte load per lane.
__device__ __forceinline__ v16bf b_frag(const u16* packed, int kc, int KC) {
  int lane = threadIdx.x & 31;
  int nt   = threadIdx.x >> 5;
  return *(const v16bf*)(packed + (((size_t)(nt * KC + kc) * 32 + lane) << 4));
}

// One GEMM stage over RT row-tiles (16 rows each); each loaded B feeds RT WMMAs.
template<int KC, int RT>
__device__ __forceinline__ void mlp_stage_bf16(const u16* aT, const u16* wP,
                                               const float* bias, u16* outT, bool act) {
  v8f acc[RT];
#pragma unroll
  for (int t = 0; t < RT; ++t)
#pragma unroll
    for (int i = 0; i < 8; ++i) acc[t][i] = 0.f;
#pragma unroll
  for (int kc = 0; kc < KC; ++kc) {
    v16bf b = b_frag(wP, kc, KC);
#pragma unroll
    for (int t = 0; t < RT; ++t) {
      v16bf a = a_frag(aT + t * 16 * KC * 32, KC * 32, kc);
      acc[t] = __builtin_amdgcn_wmma_f32_16x16x32_bf16(false, a, false, b, (short)0,
                                                       acc[t], false, false);
    }
  }
  int lane = threadIdx.x & 31, nt = threadIdx.x >> 5;
  int col = (nt << 4) + (lane & 15);
  int rb  = (lane >> 4) << 3;        // D layout: VGPR q -> row q + 8*(lane>>4)
  float bia = bias[col];
#pragma unroll
  for (int t = 0; t < RT; ++t)
#pragma unroll
    for (int q = 0; q < 8; ++q) {
      float x = acc[t][q] + bia;
      if (act) x = silu(x);
      outT[(t * 16 + rb + q) * HID + col] = f2bf(x);
    }
}

template<int KC, int RT>
__device__ __forceinline__ void mlp_stage_f32(const u16* aT, const u16* wP,
                                              const float* bias, float* outT) {
  v8f acc[RT];
#pragma unroll
  for (int t = 0; t < RT; ++t)
#pragma unroll
    for (int i = 0; i < 8; ++i) acc[t][i] = 0.f;
#pragma unroll
  for (int kc = 0; kc < KC; ++kc) {
    v16bf b = b_frag(wP, kc, KC);
#pragma unroll
    for (int t = 0; t < RT; ++t) {
      v16bf a = a_frag(aT + t * 16 * KC * 32, KC * 32, kc);
      acc[t] = __builtin_amdgcn_wmma_f32_16x16x32_bf16(false, a, false, b, (short)0,
                                                       acc[t], false, false);
    }
  }
  int lane = threadIdx.x & 31, nt = threadIdx.x >> 5;
  int col = (nt << 4) + (lane & 15);
  int rb  = (lane >> 4) << 3;
  float bia = bias[col];
#pragma unroll
  for (int t = 0; t < RT; ++t)
#pragma unroll
    for (int q = 0; q < 8; ++q)
      outT[(t * 16 + rb + q) * HID + col] = acc[t][q] + bia;
}

// LayerNorm stats, 16-row tile: 256 threads = 16 rows x 16 partials of 8
__device__ __forceinline__ void ln_stats16(const float* fT, float* red, float* mu, float* rs) {
  int tid = threadIdx.x;
  int r = tid >> 4, l = tid & 15;
  float s = 0.f, q = 0.f;
  const float* row = fT + r * HID + l * 8;
#pragma unroll
  for (int c = 0; c < 8; ++c) { float v = row[c]; s += v; q += v * v; }
  red[(r * 16 + l) * 2 + 0] = s;
  red[(r * 16 + l) * 2 + 1] = q;
  __syncthreads();
  if (tid < 16) {
    float S = 0.f, Q = 0.f;
#pragma unroll
    for (int j = 0; j < 16; ++j) { S += red[(tid * 16 + j) * 2]; Q += red[(tid * 16 + j) * 2 + 1]; }
    float m = S * (1.f / HID);
    float var = Q * (1.f / HID) - m * m;
    mu[tid] = m;
    rs[tid] = rsqrtf(var + 1e-5f);
  }
  __syncthreads();
}

// LayerNorm stats, 32-row tile: 256 threads = 32 rows x 8 partials of 16
__device__ __forceinline__ void ln_stats32(const float* fT, float* red, float* mu, float* rs) {
  int tid = threadIdx.x;
  int r = tid >> 3, l = tid & 7;
  float s = 0.f, q = 0.f;
  const float* row = fT + r * HID + l * 16;
#pragma unroll
  for (int c = 0; c < 16; ++c) { float v = row[c]; s += v; q += v * v; }
  red[(r * 8 + l) * 2 + 0] = s;
  red[(r * 8 + l) * 2 + 1] = q;
  __syncthreads();
  if (tid < 32) {
    float S = 0.f, Q = 0.f;
#pragma unroll
    for (int j = 0; j < 8; ++j) { S += red[(tid * 8 + j) * 2]; Q += red[(tid * 8 + j) * 2 + 1]; }
    float m = S * (1.f / HID);
    float var = Q * (1.f / HID) - m * m;
    mu[tid] = m;
    rs[tid] = rsqrtf(var + 1e-5f);
  }
  __syncthreads();
}

// ---------- small utility kernels ----------
__global__ void pack_w_kernel(const float* __restrict__ W, u16* __restrict__ out,
                              int K, int KC) {
  int i = blockIdx.x * 256 + threadIdx.x;
  if (i >= KC * 4096) return;
  int e = i & 15;
  int t = i >> 4;
  int lane = t & 31; t >>= 5;
  int kc = t % KC;
  int nt = t / KC;
  int k   = kc * 32 + ((lane >> 4) << 3) + (e & 7) + ((e >> 3) << 4);
  int col = (nt << 4) + (lane & 15);
  float v = (k < K) ? W[(size_t)k * HID + col] : 0.f;
  out[i] = f2bf(v);
}

__global__ void f32_to_bf16_kernel(const float* __restrict__ in, u16* __restrict__ out, int n) {
  int i = blockIdx.x * 256 + threadIdx.x;
  if (i < n) out[i] = f2bf(in[i]);
}
__global__ void copy_f32_kernel(const float* __restrict__ in, float* __restrict__ out, int n) {
  int i = blockIdx.x * 256 + threadIdx.x;
  if (i < n) out[i] = in[i];
}
__global__ void zero_f32_kernel(float* __restrict__ p, int n) {
  int i = blockIdx.x * 256 + threadIdx.x;
  if (i < n) p[i] = 0.f;
}

// ---------- edge encoder: [E,8] -> [E,128] bf16, 32 edges per block ----------
__global__ __launch_bounds__(256) void edge_enc_kernel(
    const float* __restrict__ attr, u16* __restrict__ edge_bf,
    const u16* w1, const float* b1, const u16* w2, const float* b2,
    const u16* w3, const float* b3, const float* g, const float* bb) {
  __shared__ __align__(16) u16 A[32 * 32];
  __shared__ __align__(16) u16 H1[32 * HID];
  __shared__ __align__(16) u16 H2[32 * HID];
  __shared__ float F[32 * HID];
  __shared__ float RED[32 * 8 * 2];
  __shared__ float MU[32], RS[32];
  int e0 = blockIdx.x * 32, tid = threadIdx.x;
  for (int i = tid; i < 32 * 32; i += 256) {
    int r = i >> 5, k = i & 31;
    A[i] = f2bf(k < EDIM ? attr[(size_t)(e0 + r) * EDIM + k] : 0.f);
  }
  __syncthreads();
  mlp_stage_bf16<1, 2>(A, w1, b1, H1, true);
  __syncthreads();
  mlp_stage_bf16<4, 2>(H1, w2, b2, H2, true);
  __syncthreads();
  mlp_stage_f32<4, 2>(H2, w3, b3, F);
  __syncthreads();
  ln_stats32(F, RED, MU, RS);
  int r = tid >> 3, c0 = (tid & 7) << 4;
  float m = MU[r], s = RS[r];
#pragma unroll
  for (int c = 0; c < 16; ++c) {
    int col = c0 + c;
    float v = (F[r * HID + col] - m) * s * g[col] + bb[col];
    edge_bf[(size_t)(e0 + r) * HID + col] = f2bf(v);
  }
}

// ---------- edge MLP + in-place residual + scatter-sum, 32 edges per block ----------
__global__ __launch_bounds__(256) void edge_layer_kernel(
    const u16* __restrict__ xdst_bf, const u16* __restrict__ xsrc_bf,
    u16* __restrict__ edge_bf,
    const int* __restrict__ src_idx, const int* __restrict__ dst_idx,
    const u16* w1, const float* b1, const u16* w2, const float* b2,
    const u16* w3, const float* b3, const float* g, const float* bb,
    float* __restrict__ outAcc) {
  __shared__ __align__(16) u16 A[32 * 384];
  __shared__ __align__(16) u16 H1[32 * HID];
  __shared__ __align__(16) u16 H2[32 * HID];
  __shared__ float F[32 * HID];
  __shared__ float RED[32 * 8 * 2];
  __shared__ float MU[32], RS[32];
  __shared__ int DIDX[32], SIDX[32];

  int e0 = blockIdx.x * 32;
  int tid = threadIdx.x;
  if (tid < 32) { DIDX[tid] = dst_idx[e0 + tid]; SIDX[tid] = src_idx[e0 + tid]; }
  __syncthreads();
  // Stage A = [x_dst[d] | x_src[s] | edge_attr]: async global->LDS, 16B per lane
  for (int i = tid; i < 32 * 48; i += 256) {
    int r = i / 48, q = i % 48;
    const u16* gp;
    if (q < 16)      gp = xdst_bf + (size_t)DIDX[r] * HID + q * 8;
    else if (q < 32) gp = xsrc_bf + (size_t)SIDX[r] * HID + (q - 16) * 8;
    else             gp = edge_bf + (size_t)(e0 + r) * HID + (q - 32) * 8;
    async_cp16(gp, A + r * 384 + q * 8);
  }
  async_wait();
  __syncthreads();
  mlp_stage_bf16<12, 2>(A, w1, b1, H1, true);
  __syncthreads();
  mlp_stage_bf16<4, 2>(H1, w2, b2, H2, true);
  __syncthreads();
  mlp_stage_f32<4, 2>(H2, w3, b3, F);
  __syncthreads();
  ln_stats32(F, RED, MU, RS);
  int r = tid >> 3, c0 = (tid & 7) << 4;
  int d = DIDX[r];
  float m = MU[r], s = RS[r];
#pragma unroll
  for (int c = 0; c < 16; ++c) {
    int col = c0 + c;
    float resid = bf2f(A[r * 384 + 256 + col]);         // original edge_attr
    float v = (F[r * HID + col] - m) * s * g[col] + bb[col] + resid;
    edge_bf[(size_t)(e0 + r) * HID + col] = f2bf(v);    // in-place edges_new
    atomic_add_f32(outAcc + (size_t)d * HID + col, v);  // segment_sum
  }
}

// ---------- node MLP + residual (f32 master in d_out, bf16 mirror), 16 rows ----------
__global__ __launch_bounds__(256) void node_layer_kernel(
    const float* __restrict__ outAcc, float* __restrict__ xdst_f,
    u16* __restrict__ xdst_bf,
    const u16* w1, const float* b1, const u16* w2, const float* b2,
    const u16* w3, const float* b3, const float* g, const float* bb) {
  __shared__ __align__(16) u16 A[16 * 256];
  __shared__ __align__(16) u16 H1[16 * HID];
  __shared__ __align__(16) u16 H2[16 * HID];
  __shared__ float F[16 * HID];
  __shared__ float RED[16 * 16 * 2];
  __shared__ float MU[16], RS[16];
  int n0 = blockIdx.x * 16;
  int tid = threadIdx.x;
  {
    // x_dst rows: async global->LDS (256 lanes x 16B covers 16x128 bf16)
    int r = tid >> 4, q = tid & 15;
    async_cp16(xdst_bf + (size_t)(n0 + r) * HID + q * 8, A + r * 256 + q * 8);
  }
  for (int i = tid; i < 16 * HID; i += 256) {
    int r = i >> 7, c = i & 127;
    A[r * 256 + 128 + c] = f2bf(outAcc[(size_t)(n0 + r) * HID + c]);
  }
  async_wait();
  __syncthreads();
  mlp_stage_bf16<8, 1>(A, w1, b1, H1, true);
  __syncthreads();
  mlp_stage_bf16<4, 1>(H1, w2, b2, H2, true);
  __syncthreads();
  mlp_stage_f32<4, 1>(H2, w3, b3, F);
  __syncthreads();
  ln_stats16(F, RED, MU, RS);
  int r = tid >> 4, c0 = (tid & 15) << 3;
  float m = MU[r], s = RS[r];
#pragma unroll
  for (int c = 0; c < 8; ++c) {
    int col = c0 + c;
    size_t gi = (size_t)(n0 + r) * HID + col;
    float v = (F[r * HID + col] - m) * s * g[col] + bb[col] + xdst_f[gi];
    xdst_f[gi] = v;
    xdst_bf[gi] = f2bf(v);
  }
}

// ---------- host launch ----------
extern "C" void kernel_launch(void* const* d_in, const int* in_sizes, int n_in,
                              void* d_out, int out_size, void* d_ws, size_t ws_size,
                              hipStream_t stream) {
  (void)in_sizes; (void)n_in; (void)out_size; (void)ws_size;
  const float* x_src  = (const float*)d_in[0];
  const float* x_dst  = (const float*)d_in[1];
  const int*   eidx   = (const int*)d_in[2];
  const float* eattr  = (const float*)d_in[3];
  const int* src_idx  = eidx;
  const int* dst_idx  = eidx + NEDGE;

  char* base = (char*)d_ws;
  size_t off = 0;
  auto carve = [&](size_t bytes) -> void* {
    void* p = base + off;
    off += (bytes + 255) & ~(size_t)255;
    return p;
  };
  u16*   edge_bf = (u16*)carve((size_t)NEDGE * HID * 2);
  float* outAcc  = (float*)carve((size_t)NDST * HID * 4);
  u16*   xsrc_bf = (u16*)carve((size_t)NSRC * HID * 2);
  u16*   xdst_bf = (u16*)carve((size_t)NDST * HID * 2);

  auto packW = [&](int argi, int K) -> u16* {
    int KC = (K + 31) >> 5;
    u16* dst = (u16*)carve((size_t)KC * 4096 * 2);
    int n = KC * 4096;
    pack_w_kernel<<<(n + 255) / 256, 256, 0, stream>>>((const float*)d_in[argi], dst, K, KC);
    return dst;
  };

  // params flatten order (insertion order): edge_enc {l1.w,l1.b,l2.w,l2.b,l3.w,l3.b,ln_g,ln_b}
  // then blocks[i] { node_mlp(8 arrays), edge_mlp(8 arrays) }
  u16* encW[3]; const float* encB[3];
  encW[0] = packW(4, EDIM); encB[0] = (const float*)d_in[5];
  encW[1] = packW(6, HID);  encB[1] = (const float*)d_in[7];
  encW[2] = packW(8, HID);  encB[2] = (const float*)d_in[9];
  const float* encG  = (const float*)d_in[10];
  const float* encBB = (const float*)d_in[11];

  u16* nW[NLAYERS][3]; u16* eW[NLAYERS][3];
  const float *nB[NLAYERS][3], *eB[NLAYERS][3];
  const float *nG[NLAYERS], *nBB[NLAYERS], *eG[NLAYERS], *eBB[NLAYERS];
  for (int i = 0; i < NLAYERS; ++i) {
    int b = 12 + 16 * i;
    nW[i][0] = packW(b + 0, 2 * HID);  nB[i][0] = (const float*)d_in[b + 1];
    nW[i][1] = packW(b + 2, HID);      nB[i][1] = (const float*)d_in[b + 3];
    nW[i][2] = packW(b + 4, HID);      nB[i][2] = (const float*)d_in[b + 5];
    nG[i] = (const float*)d_in[b + 6]; nBB[i] = (const float*)d_in[b + 7];
    eW[i][0] = packW(b + 8, 3 * HID);  eB[i][0] = (const float*)d_in[b + 9];
    eW[i][1] = packW(b + 10, HID);     eB[i][1] = (const float*)d_in[b + 11];
    eW[i][2] = packW(b + 12, HID);     eB[i][2] = (const float*)d_in[b + 13];
    eG[i] = (const float*)d_in[b + 14]; eBB[i] = (const float*)d_in[b + 15];
  }

  f32_to_bf16_kernel<<<(NSRC * HID + 255) / 256, 256, 0, stream>>>(x_src, xsrc_bf, NSRC * HID);
  f32_to_bf16_kernel<<<(NDST * HID + 255) / 256, 256, 0, stream>>>(x_dst, xdst_bf, NDST * HID);
  copy_f32_kernel<<<(NDST * HID + 255) / 256, 256, 0, stream>>>(x_dst, (float*)d_out, NDST * HID);

  edge_enc_kernel<<<NEDGE / 32, 256, 0, stream>>>(eattr, edge_bf,
      encW[0], encB[0], encW[1], encB[1], encW[2], encB[2], encG, encBB);

  for (int i = 0; i < NLAYERS; ++i) {
    zero_f32_kernel<<<(NDST * HID + 255) / 256, 256, 0, stream>>>(outAcc, NDST * HID);
    edge_layer_kernel<<<NEDGE / 32, 256, 0, stream>>>(xdst_bf, xsrc_bf, edge_bf,
        src_idx, dst_idx,
        eW[i][0], eB[i][0], eW[i][1], eB[i][1], eW[i][2], eB[i][2], eG[i], eBB[i],
        outAcc);
    node_layer_kernel<<<NDST / 16, 256, 0, stream>>>(outAcc, (float*)d_out, xdst_bf,
        nW[i][0], nB[i][0], nW[i][1], nB[i][1], nW[i][2], nB[i][2], nG[i], nBB[i]);
  }
}
